// WindowAttention_94489281360
// MI455X (gfx1250) — matrix operand order repaired and checked by reading, not compiled, not run on hardware
//
#include <hip/hip_runtime.h>

// ---------------------------------------------------------------------------
// SwinV2 window attention for MI455X (gfx1250), wave32 + WMMA bf16.
// Shapes: B_=512 windows, N=144 tokens, C=512, H=16 heads, DH=32, NW=64 masks.
// ---------------------------------------------------------------------------

typedef __attribute__((ext_vector_type(16))) __bf16 bf16x16;
typedef __attribute__((ext_vector_type(8)))  __bf16 bf16x8;
typedef __attribute__((ext_vector_type(4)))  __bf16 bf16x4;
typedef __attribute__((ext_vector_type(8)))  float  v8f;

#define NTOK   144
#define NHEAD  16
#define DHEAD  32
#define CDIM   512
#define BWIN   512
#define NWMASK 64
#define MROWS  (BWIN * NTOK)          // 73728 token rows

__device__ __forceinline__ v8f zero8() {
    v8f z;
    #pragma unroll
    for (int i = 0; i < 8; ++i) z[i] = 0.0f;
    return z;
}

__device__ __forceinline__ bf16x16 join8(bf16x8 lo, bf16x8 hi) {
    bf16x16 r;
    #pragma unroll
    for (int i = 0; i < 8; ++i) { r[i] = lo[i]; r[8 + i] = hi[i]; }
    return r;
}

// -------------------- Stage 1: CPB MLP table (529 x 16) --------------------
__global__ __launch_bounds__(128)
void cpb_table_kernel(const float* __restrict__ rel_table,
                      const float* __restrict__ w1, const float* __restrict__ b1,
                      const float* __restrict__ w2, float* __restrict__ tbl) {
    int p = blockIdx.x;                 // 0..528
    int tid = threadIdx.x;              // 128 threads
    float t0 = rel_table[p * 2 + 0];
    float t1 = rel_table[p * 2 + 1];
    __shared__ float red[16][128];
    float part[16];
    #pragma unroll
    for (int h = 0; h < 16; ++h) part[h] = 0.0f;
    for (int j = tid; j < 512; j += 128) {
        float hv = fmaxf(w1[j * 2] * t0 + w1[j * 2 + 1] * t1 + b1[j], 0.0f);
        #pragma unroll
        for (int h = 0; h < 16; ++h) part[h] += w2[h * 512 + j] * hv;
    }
    #pragma unroll
    for (int h = 0; h < 16; ++h) red[h][tid] = part[h];
    __syncthreads();
    for (int s = 64; s > 0; s >>= 1) {
        if (tid < s) {
            #pragma unroll
            for (int h = 0; h < 16; ++h) red[h][tid] += red[h][tid + s];
        }
        __syncthreads();
    }
    if (tid < 16) tbl[p * 16 + tid] = red[tid][0];
}

// -------------- Stage 2: gather + 16*sigmoid -> bias (H,N,N) ---------------
__global__ __launch_bounds__(256)
void bias_expand_kernel(const float* __restrict__ tbl,
                        const int* __restrict__ rel_index,
                        float* __restrict__ biasb) {
    int idx = blockIdx.x * 256 + threadIdx.x;   // H*N*N = 331776 exactly
    int h  = idx / (NTOK * NTOK);
    int ij = idx - h * (NTOK * NTOK);
    float v = tbl[rel_index[ij] * 16 + h];
    biasb[idx] = 16.0f / (1.0f + __expf(-v));
}

// ---------------- bf16 WMMA GEMM: C = A(MxK) * W(NcxK)^T -------------------
// Block 256 threads (8 waves), tile 128(M) x 128(N), BK=32.
// Waves arranged 2(M) x 4(N); per-wave tile 64x32 -> 4x2 16x16 frags.
// Double-buffered LDS, one barrier per K-step; next tile staged in registers.
// Grid: (Nc/128, M/128) so concurrent blocks share one A-stripe in L2.
__device__ __forceinline__ void gload4(const float* s, float* d) {
    float4 v = *reinterpret_cast<const float4*>(s);
    d[0] = v.x; d[1] = v.y; d[2] = v.z; d[3] = v.w;
}
__device__ __forceinline__ void gload4(const __bf16* s, __bf16* d) {
    bf16x4 v = *reinterpret_cast<const bf16x4*>(s);
    d[0] = v[0]; d[1] = v[1]; d[2] = v[2]; d[3] = v[3];
}
__device__ __forceinline__ __bf16 tobf(float f)  { return (__bf16)f; }
__device__ __forceinline__ __bf16 tobf(__bf16 f) { return f; }

#define GSTR 40                 // LDS row stride in halves (80B, 16B aligned)
#define GBUF (128 * GSTR)       // one A or B buffer (halves)

template <typename AT, bool OUT_F32>
__global__ __launch_bounds__(256)
void gemm_bf16_wmma(const AT* __restrict__ A, const float* __restrict__ W,
                    const float* __restrict__ bias, void* __restrict__ out,
                    int M, int Nc, int K) {
    __shared__ __align__(16) __bf16 As[2 * GBUF];
    __shared__ __align__(16) __bf16 Bs[2 * GBUF];
    int tid  = threadIdx.x;
    int lane = tid & 31;
    int wv   = tid >> 5;          // 0..7
    int wm   = wv & 1;            // 2 waves along M (64 rows each)
    int wn   = wv >> 1;           // 4 waves along N (32 cols each)
    int n0   = blockIdx.x * 128;
    int m0   = blockIdx.y * 128;
    int mlo  = lane & 15;
    int hi   = lane >> 4;         // 0/1

    v8f acc[4][2];
    #pragma unroll
    for (int a = 0; a < 4; ++a)
        #pragma unroll
        for (int b = 0; b < 2; ++b) acc[a][b] = zero8();

    AT    aReg[4][4];
    float bReg[4][4];
    int   lrow[4], lcol[4];
    #pragma unroll
    for (int i = 0; i < 4; ++i) {
        int idx = tid + i * 256;           // 0..1023 over 128x(32/4) vec4 slots
        lrow[i] = idx >> 3;
        lcol[i] = (idx & 7) * 4;
    }

    const int nk = K / 32;

    // stage K-tile into registers
    auto loadTiles = [&](int k0) {
        #pragma unroll
        for (int i = 0; i < 4; ++i) {
            gload4(A + (size_t)(m0 + lrow[i]) * K + k0 + lcol[i], aReg[i]);
            gload4(W + (size_t)(n0 + lrow[i]) * K + k0 + lcol[i], bReg[i]);
        }
    };
    // commit staged registers to LDS buffer `buf`
    auto storeTiles = [&](int buf) {
        #pragma unroll
        for (int i = 0; i < 4; ++i) {
            __bf16* da = As + buf * GBUF + lrow[i] * GSTR + lcol[i];
            __bf16* db = Bs + buf * GBUF + lrow[i] * GSTR + lcol[i];
            #pragma unroll
            for (int j = 0; j < 4; ++j) { da[j] = tobf(aReg[i][j]); db[j] = tobf(bReg[i][j]); }
        }
    };

    loadTiles(0);
    storeTiles(0);

    for (int it = 0; it < nk; ++it) {
        __syncthreads();                       // buf[it&1] visible to all waves
        if (it + 2 < nk) {                     // gfx1250 L2 prefetch of K+2 tile
            __builtin_prefetch(A + (size_t)(m0 + lrow[0]) * K + (it + 2) * 32, 0, 1);
            __builtin_prefetch(W + (size_t)(n0 + lrow[0]) * K + (it + 2) * 32, 0, 1);
        }
        if (it + 1 < nk) loadTiles((it + 1) * 32);   // overlap with WMMAs below

        const __bf16* ab = As + (it & 1) * GBUF;
        const __bf16* bb = Bs + (it & 1) * GBUF;
        bf16x16 af[4], bfm[2];
        #pragma unroll
        for (int mt = 0; mt < 4; ++mt) {
            const __bf16* p = ab + (wm * 64 + mt * 16 + mlo) * GSTR + hi * 8;
            af[mt] = join8(*(const bf16x8*)p, *(const bf16x8*)(p + 16));
        }
        #pragma unroll
        for (int nt = 0; nt < 2; ++nt) {
            const __bf16* p = bb + (wn * 32 + nt * 16 + mlo) * GSTR + hi * 8;
            bfm[nt] = join8(*(const bf16x8*)p, *(const bf16x8*)(p + 16));
        }
        #pragma unroll
        for (int mt = 0; mt < 4; ++mt)
            #pragma unroll
            for (int nt = 0; nt < 2; ++nt)
                acc[mt][nt] = __builtin_amdgcn_wmma_f32_16x16x32_bf16(
                    false, af[mt], false, bfm[nt], (short)0, acc[mt][nt], false, false);

        if (it + 1 < nk) storeTiles((it + 1) & 1);   // fill other buffer
    }

    #pragma unroll
    for (int mt = 0; mt < 4; ++mt)
        #pragma unroll
        for (int nt = 0; nt < 2; ++nt)
            #pragma unroll
            for (int r = 0; r < 8; ++r) {
                int gm = m0 + wm * 64 + mt * 16 + hi * 8 + r;
                int gn = n0 + wn * 32 + nt * 16 + mlo;
                float v = acc[mt][nt][r];
                if constexpr (OUT_F32)
                    ((float*)out)[(size_t)gm * Nc + gn] = v + bias[gn];
                else
                    ((__bf16*)out)[(size_t)gm * Nc + gn] = (__bf16)v;
            }
}

// ---------------- Stage 4: attention core, 1 block per (b,h) ---------------
#define KSTR 40      // K-tile LDS row stride (halves), 80B = 16B aligned
#define VSTR 168     // V^T LDS row stride (halves), pads K to >=160 for PV
#define PSTR 160     // P LDS row stride (halves)

__global__ __launch_bounds__(288)
void attn_kernel(const __bf16* __restrict__ qkv,     // (M,1536) bf16
                 const float* __restrict__ biasb,    // (H,N,N)
                 const float* __restrict__ mask,     // (NW,N,N)
                 const float* __restrict__ logit_scale, // (H)
                 __bf16* __restrict__ attn_o) {      // (M,512) bf16
    __shared__ __align__(16) __bf16 kn[NTOK * KSTR];
    __shared__ __align__(16) __bf16 vt[DHEAD * VSTR];
    __shared__ __align__(16) __bf16 pl[9 * 16 * PSTR];

    int bh = blockIdx.x;
    int b = bh >> 4, h = bh & 15;
    int tid = threadIdx.x;
    int lane = tid & 31;
    int wv = tid >> 5;                 // 0..8 query tiles
    int mlo = lane & 15;
    int hi = lane >> 4;
    size_t rowbase = (size_t)b * NTOK;

    float scale = __expf(fminf(logit_scale[h], 4.605170185988092f)); // log(100)

    // ---- load K rows (normalize) and V rows (transpose) into LDS ----
    if (tid < NTOK) {
        const bf16x8* kr = (const bf16x8*)(qkv + (rowbase + tid) * 1536 + 512 + h * 32);
        float tmp[32]; float ss = 0.0f;
        #pragma unroll
        for (int c = 0; c < 4; ++c) {
            bf16x8 v = kr[c];
            #pragma unroll
            for (int j = 0; j < 8; ++j) { float f = (float)v[j]; tmp[c * 8 + j] = f; ss += f * f; }
        }
        float rn = rsqrtf(ss + 1e-12f);
        #pragma unroll
        for (int d = 0; d < 32; ++d) kn[tid * KSTR + d] = (__bf16)(tmp[d] * rn);

        const bf16x8* vr = (const bf16x8*)(qkv + (rowbase + tid) * 1536 + 1024 + h * 32);
        #pragma unroll
        for (int c = 0; c < 4; ++c) {
            bf16x8 v = vr[c];
            #pragma unroll
            for (int j = 0; j < 8; ++j) vt[(c * 8 + j) * VSTR + tid] = v[j];
        }
    }
    // zero V^T pad columns (k = 144..VSTR-1) so padded-P dot products stay finite
    for (int idx = tid; idx < DHEAD * (VSTR - NTOK); idx += 288) {
        int d = idx / (VSTR - NTOK), k = NTOK + idx % (VSTR - NTOK);
        vt[d * VSTR + k] = (__bf16)0.0f;
    }
    __syncthreads();

    // ---- build normalized Q A-fragment for this wave's 16-row tile ----
    bf16x16 afrag;
    {
        const __bf16* qr = qkv + (rowbase + wv * 16 + mlo) * 1536 + h * 32;
        bf16x8 q0 = *(const bf16x8*)(qr + hi * 8);
        bf16x8 q1 = *(const bf16x8*)(qr + 16 + hi * 8);
        float f0[8], f1[8], ss = 0.0f;
        #pragma unroll
        for (int i = 0; i < 8; ++i) {
            f0[i] = (float)q0[i]; f1[i] = (float)q1[i];
            ss += f0[i] * f0[i] + f1[i] * f1[i];
        }
        ss += __shfl_xor(ss, 16, 32);              // other half of this row
        float rn = rsqrtf(ss + 1e-12f);
        #pragma unroll
        for (int i = 0; i < 8; ++i) {
            afrag[i]     = (__bf16)(f0[i] * rn);
            afrag[8 + i] = (__bf16)(f1[i] * rn);
        }
    }

    const float* bp = biasb + (size_t)h * NTOK * NTOK + (size_t)(wv * 16) * NTOK;
    const float* mp = mask + (size_t)(b & (NWMASK - 1)) * NTOK * NTOK + (size_t)(wv * 16) * NTOK;

    // ---- S = (qn @ kn^T)*scale + bias + mask : 9 WMMAs ----
    float S[9][8];
    #pragma unroll
    for (int kt = 0; kt < 9; ++kt) {
        const __bf16* kp = kn + (kt * 16 + mlo) * KSTR + hi * 8;
        bf16x16 bfrag = join8(*(const bf16x8*)kp, *(const bf16x8*)(kp + 16));
        v8f c = __builtin_amdgcn_wmma_f32_16x16x32_bf16(
            false, afrag, false, bfrag, (short)0, zero8(), false, false);
        int kc = kt * 16 + mlo;
        #pragma unroll
        for (int r = 0; r < 8; ++r) {
            int q = hi * 8 + r;
            S[kt][r] = c[r] * scale + bp[q * NTOK + kc] + mp[q * NTOK + kc];
        }
    }

    // ---- softmax over 144 keys (half-wave shuffles: rows live in 16 lanes) ----
    #pragma unroll
    for (int r = 0; r < 8; ++r) {
        float m = S[0][r];
        #pragma unroll
        for (int kt = 1; kt < 9; ++kt) m = fmaxf(m, S[kt][r]);
        #pragma unroll
        for (int off = 8; off >= 1; off >>= 1) m = fmaxf(m, __shfl_xor(m, off, 32));
        float s = 0.0f;
        #pragma unroll
        for (int kt = 0; kt < 9; ++kt) { float e = __expf(S[kt][r] - m); S[kt][r] = e; s += e; }
        #pragma unroll
        for (int off = 8; off >= 1; off >>= 1) s += __shfl_xor(s, off, 32);
        float inv = 1.0f / s;
        #pragma unroll
        for (int kt = 0; kt < 9; ++kt) S[kt][r] *= inv;
    }

    // ---- P -> LDS (bf16, zero-padded to 160 keys), wave-private region ----
    __bf16* pw = pl + wv * 16 * PSTR;
    #pragma unroll
    for (int kt = 0; kt < 9; ++kt)
        #pragma unroll
        for (int r = 0; r < 8; ++r)
            pw[(hi * 8 + r) * PSTR + kt * 16 + mlo] = (__bf16)S[kt][r];
    #pragma unroll
    for (int r = 0; r < 8; ++r)
        pw[(hi * 8 + r) * PSTR + NTOK + mlo] = (__bf16)0.0f;   // cols 144..159

    // ---- O = P @ V : 2 output tiles x 5 K-chunks of WMMAs ----
    #pragma unroll
    for (int dt = 0; dt < 2; ++dt) {
        v8f acc = zero8();
        #pragma unroll
        for (int kc = 0; kc < 5; ++kc) {
            const __bf16* pp = pw + mlo * PSTR + kc * 32 + hi * 8;
            bf16x16 pa = join8(*(const bf16x8*)pp, *(const bf16x8*)(pp + 16));
            const __bf16* vp = vt + (dt * 16 + mlo) * VSTR + kc * 32 + hi * 8;
            bf16x16 vb = join8(*(const bf16x8*)vp, *(const bf16x8*)(vp + 16));
            acc = __builtin_amdgcn_wmma_f32_16x16x32_bf16(
                false, pa, false, vb, (short)0, acc, false, false);
        }
        #pragma unroll
        for (int r = 0; r < 8; ++r) {
            size_t row = rowbase + wv * 16 + hi * 8 + r;
            attn_o[row * CDIM + h * 32 + dt * 16 + mlo] = (__bf16)acc[r];
        }
    }
}

// ---------------------------------------------------------------------------
extern "C" void kernel_launch(void* const* d_in, const int* in_sizes, int n_in,
                              void* d_out, int out_size, void* d_ws, size_t ws_size,
                              hipStream_t stream) {
    const float* x           = (const float*)d_in[0];
    const float* mask        = (const float*)d_in[1];
    const float* qkv_w       = (const float*)d_in[2];
    const float* logit_scale = (const float*)d_in[3];
    const float* cpb_w1      = (const float*)d_in[4];
    const float* cpb_b1      = (const float*)d_in[5];
    const float* cpb_w2      = (const float*)d_in[6];
    const float* proj_w      = (const float*)d_in[7];
    const float* proj_b      = (const float*)d_in[8];
    const float* rel_table   = (const float*)d_in[9];
    const int*   rel_index   = (const int*)d_in[10];

    char* ws = (char*)d_ws;
    __bf16* qkv    = (__bf16*)ws;  ws += (size_t)MROWS * 1536 * sizeof(__bf16);
    __bf16* attn_o = (__bf16*)ws;  ws += (size_t)MROWS * CDIM * sizeof(__bf16);
    float*  tbl    = (float*)ws;   ws += (size_t)529 * 16 * sizeof(float);
    float*  biasb  = (float*)ws;   // H*N*N floats

    // 1) CPB MLP table (529 positions)
    cpb_table_kernel<<<529, 128, 0, stream>>>(rel_table, cpb_w1, cpb_b1, cpb_w2, tbl);
    // 2) gather + 16*sigmoid -> (H,N,N) bias
    bias_expand_kernel<<<(NHEAD * NTOK * NTOK) / 256, 256, 0, stream>>>(tbl, rel_index, biasb);
    // 3) QKV GEMM: (73728 x 512) x (512 x 1536) -> bf16; N-tiles fastest for L2 reuse
    gemm_bf16_wmma<float, false><<<dim3(1536 / 128, MROWS / 128), 256, 0, stream>>>(
        x, qkv_w, nullptr, qkv, MROWS, 1536, CDIM);
    // 4) cosine attention + bias + mask + softmax + PV, per (window, head)
    attn_kernel<<<BWIN * NHEAD, 288, 0, stream>>>(qkv, biasb, mask, logit_scale, attn_o);
    // 5) output projection: (73728 x 512) x (512 x 512) + bias -> f32 d_out
    gemm_bf16_wmma<__bf16, true><<<dim3(CDIM / 128, MROWS / 128), 256, 0, stream>>>(
        attn_o, proj_w, proj_b, d_out, MROWS, CDIM, CDIM);
}
